// DegreeLevelRatePredictor_44401371906118
// MI455X (gfx1250) — compile-verified, matrix operand order checked by reading, not run on hardware
//
#include <hip/hip_runtime.h>
#include <hip/hip_bf16.h>

// ---------------------------------------------------------------------------
// DegreeLevelRatePredictor forward pass for gfx1250 (MI455X), wave32 + WMMA.
// B=16, S=1024, K=100 (101 classes), H=128, NH=4 (hd=32), NL=2.
// ---------------------------------------------------------------------------

typedef __attribute__((ext_vector_type(16))) _Float16 v16h;
typedef __attribute__((ext_vector_type(8)))  _Float16 v8h;
typedef __attribute__((ext_vector_type(8)))  float    v8f;

#define BSEQ   16
#define SEQ    1024
#define HDIM   128
#define NHEAD  4
#define HD     32
#define NLAYER 2
#define NCLS   101
#define ROWS   (BSEQ * SEQ)      // 16384

// LDS tile stride in halves: 32 data + 8 pad = 40 halves = 80 bytes (16B-aligned)
#define LSTR 40

// ---------------------------------------------------------------------------
// Load one 16x32 f16 WMMA operand fragment from an LDS tile laid out as
// tile[row][k] with `stride` halves per row (k-extent exactly 32).
// Matches CDNA5 ISA 16-bit A/B layout:
//   lanes 0-15 : row = row0+lane, halves 0-7 = k 0..7,  halves 8-15 = k 16..23
//   lanes 16-31: row = row0+lane-16, halves 0-7 = k 8..15, halves 8-15 = k 24..31
// ---------------------------------------------------------------------------
__device__ __forceinline__ v16h lds_frag(const _Float16* tile, int row0,
                                         int stride, int lane) {
  const int ll   = lane & 15;
  const int hoff = (lane >> 4) << 3;   // 0 or 8
  const _Float16* p = tile + (row0 + ll) * stride;
  union { v16h v; v8h h[2]; } u;
  u.h[0] = *(const v8h*)(p + hoff);
  u.h[1] = *(const v8h*)(p + 16 + hoff);
  return u.v;
}

__device__ __forceinline__ v8f wmma_f16(v16h a, v16h b, v8f c) {
  return __builtin_amdgcn_wmma_f32_16x16x32_f16(false, a, false, b,
                                                (short)0, c, false, false);
}

// ---------------------------------------------------------------------------
// Embedding + positional encoding.
// x[b,s,:] = W_emb[level] + (level<K ? rate*W_emb[level+1] : 0) + b_emb + pe(b)
// ---------------------------------------------------------------------------
__global__ __launch_bounds__(128) void embed_kernel(const float* __restrict__ xq,
                                                    const float* __restrict__ Wemb,
                                                    const float* __restrict__ bemb,
                                                    float* __restrict__ X) {
  const int bs = blockIdx.x;          // 0..16383
  const int t  = threadIdx.x;         // 0..127
  const int b  = bs >> 10;            // batch index (positional index!)
  const int   level = (int)xq[(size_t)bs * 2 + 0];
  const float rate  = xq[(size_t)bs * 2 + 1];
  float v = Wemb[(size_t)level * HDIM + t] + bemb[t];
  if (level < 100) v += rate * Wemb[(size_t)(level + 1) * HDIM + t];
  // sinusoidal PE over batch dim, interleaved sin/cos
  const int   i2  = (t >> 1) << 1;
  const float div = __expf((float)i2 * (-9.210340371976184f / 128.0f));
  const float ang = (float)b * div;
  v += (t & 1) ? __cosf(ang) : __sinf(ang);
  X[(size_t)bs * HDIM + t] = v;
}

// ---------------------------------------------------------------------------
// WMMA GEMM: C[M,N] = A[M,K] @ B[K,N] + bias (+ optional ReLU)
// Compile-time K/N/RELU. Block tile 128x128, 256 threads = 8 waves,
// wave tile 32x64 (2x4 fragments, 8 WMMAs per k-step), K stepped by 32
// through padded f16 LDS tiles (conflict-free 16B ds_load_b128 reads).
// ---------------------------------------------------------------------------
template <int K, int N, bool RELU>
__global__ __launch_bounds__(256) void gemm_wmma_kernel(
    const float* __restrict__ A, const float* __restrict__ B,
    const float* __restrict__ bias, float* __restrict__ C) {
  __shared__ __align__(16) _Float16 As[128 * LSTR];   // As[m][k]
  __shared__ __align__(16) _Float16 Bs[128 * LSTR];   // Bs[n][k] (transposed)
  const int tid  = threadIdx.x;
  const int lane = tid & 31;
  const int w    = tid >> 5;
  const int m_blk = blockIdx.y * 128;
  const int n_blk = blockIdx.x * 128;
  const int wm = (w & 3)  * 32;   // 0,32,64,96
  const int wn = (w >> 2) * 64;   // 0,64

  v8f acc[2][4] = {};

  for (int k0 = 0; k0 < K; k0 += 32) {
    // A tile 128x32: coalesced on k
    #pragma unroll
    for (int i = 0; i < 16; ++i) {
      int idx = tid + i * 256;
      int r = idx >> 5, c = idx & 31;
      As[r * LSTR + c] = (_Float16)A[(size_t)(m_blk + r) * K + k0 + c];
    }
    // B tile 32x128 transposed into Bs[n][k]: coalesced on n
    #pragma unroll
    for (int i = 0; i < 16; ++i) {
      int idx = tid + i * 256;
      int kr = idx >> 7, nc = idx & 127;
      Bs[nc * LSTR + kr] = (_Float16)B[(size_t)(k0 + kr) * N + n_blk + nc];
    }
    __syncthreads();

    v16h a0 = lds_frag(As, wm,      LSTR, lane);
    v16h a1 = lds_frag(As, wm + 16, LSTR, lane);
    #pragma unroll
    for (int j = 0; j < 4; ++j) {
      v16h bf = lds_frag(Bs, wn + j * 16, LSTR, lane);
      acc[0][j] = wmma_f16(a0, bf, acc[0][j]);
      acc[1][j] = wmma_f16(a1, bf, acc[1][j]);
    }
    __syncthreads();
  }

  // C fragment layout: vgpr r -> row base+hoff+r, col = lane&15
  const int ll   = lane & 15;
  const int hoff = (lane >> 4) << 3;
  #pragma unroll
  for (int i = 0; i < 2; ++i)
    #pragma unroll
    for (int j = 0; j < 4; ++j) {
      const int mb = m_blk + wm + i * 16 + hoff;
      const int nb = n_blk + wn + j * 16 + ll;
      const float bv = bias[nb];
      #pragma unroll
      for (int r = 0; r < 8; ++r) {
        float vv = acc[i][j][r] + bv;
        if (RELU) vv = fmaxf(vv, 0.0f);
        C[(size_t)(mb + r) * N + nb] = vv;
      }
    }
}

// ---------------------------------------------------------------------------
// Flash attention, one workgroup per (batch, head, 128-query tile).
// hd=32 == one WMMA K-step. Keys streamed 32 at a time with online softmax.
// qkv layout: [b,s, 3*128] with q at +0, k at +128, v at +256 (per head *32).
// Output att: [b,s,128] with head h occupying cols h*32..h*32+31.
// ---------------------------------------------------------------------------
__global__ __launch_bounds__(256) void attn_kernel(const float* __restrict__ qkv,
                                                   float* __restrict__ att) {
  __shared__ __align__(16) _Float16 Qs[128 * LSTR];       // Qs[qrow][hd] (scaled)
  __shared__ __align__(16) _Float16 Ks[32 * LSTR];        // Ks[key][hd]
  __shared__ __align__(16) _Float16 Vt[32 * LSTR];        // Vt[hd][key]
  __shared__ __align__(16) _Float16 Ps[8][16 * LSTR];     // per-wave P scratch
  const int tid  = threadIdx.x;
  const int lane = tid & 31;
  const int w    = tid >> 5;
  const int qb   = blockIdx.x * 128;
  const int h    = blockIdx.y;
  const int b    = blockIdx.z;
  const int ll   = lane & 15;
  const int hoff = (lane >> 4) << 3;
  const float scale = 0.17677669529663687f;  // 1/sqrt(32), folded into Q

  #pragma unroll
  for (int i = 0; i < 16; ++i) {
    int idx = tid + i * 256;
    int r = idx >> 5, c = idx & 31;
    Qs[r * LSTR + c] =
        (_Float16)(qkv[(size_t)(b * SEQ + qb + r) * 384 + h * HD + c] * scale);
  }
  __syncthreads();
  const v16h qa = lds_frag(Qs, w * 16, LSTR, lane);

  v8f o0 = {}, o1 = {};
  float mrun[8], lrun[8];
  #pragma unroll
  for (int r = 0; r < 8; ++r) { mrun[r] = -1e30f; lrun[r] = 0.0f; }

  for (int kb = 0; kb < SEQ; kb += 32) {
    __syncthreads();  // previous chunk fully consumed before overwrite
    #pragma unroll
    for (int i = 0; i < 4; ++i) {
      int idx = tid + i * 256;
      int r = idx >> 5, c = idx & 31;
      size_t base = (size_t)(b * SEQ + kb + r) * 384;
      Ks[r * LSTR + c] = (_Float16)qkv[base + 128 + h * HD + c];
      Vt[c * LSTR + r] = (_Float16)qkv[base + 256 + h * HD + c];  // transpose
    }
    __syncthreads();

    // scores: 16 queries x 32 keys (two 16x16 fragments)
    v16h kb0 = lds_frag(Ks, 0,  LSTR, lane);
    v16h kb1 = lds_frag(Ks, 16, LSTR, lane);
    v8f s0 = {}, s1 = {};
    s0 = wmma_f16(qa, kb0, s0);
    s1 = wmma_f16(qa, kb1, s1);

    // online softmax; row r lives across a 16-lane group (xor 1,2,4,8)
    #pragma unroll
    for (int r = 0; r < 8; ++r) {
      float mx = fmaxf(s0[r], s1[r]);
      #pragma unroll
      for (int off = 1; off < 16; off <<= 1)
        mx = fmaxf(mx, __shfl_xor(mx, off, 32));
      const float nm = fmaxf(mrun[r], mx);
      const float p0 = __expf(s0[r] - nm);
      const float p1 = __expf(s1[r] - nm);
      float rs = p0 + p1;
      #pragma unroll
      for (int off = 1; off < 16; off <<= 1)
        rs += __shfl_xor(rs, off, 32);
      const float alpha = __expf(mrun[r] - nm);
      lrun[r] = lrun[r] * alpha + rs;
      mrun[r] = nm;
      o0[r] *= alpha;
      o1[r] *= alpha;
      // re-layout P: C-fragment -> LDS row-major (wave-private, DS in-order)
      Ps[w][(r + hoff) * LSTR + ll]      = (_Float16)p0;
      Ps[w][(r + hoff) * LSTR + 16 + ll] = (_Float16)p1;
    }

    v16h pa  = lds_frag(Ps[w], 0, LSTR, lane);   // 16 queries x 32 keys
    v16h vb0 = lds_frag(Vt, 0,  LSTR, lane);     // hd cols 0..15
    v16h vb1 = lds_frag(Vt, 16, LSTR, lane);     // hd cols 16..31
    o0 = wmma_f16(pa, vb0, o0);
    o1 = wmma_f16(pa, vb1, o1);
  }

  #pragma unroll
  for (int r = 0; r < 8; ++r) {
    const float inv = 1.0f / lrun[r];
    const size_t row = (size_t)b * SEQ + qb + w * 16 + hoff + r;
    att[row * HDIM + h * HD + ll]      = o0[r] * inv;
    att[row * HDIM + h * HD + 16 + ll] = o1[r] * inv;
  }
}

// ---------------------------------------------------------------------------
// x = LayerNorm(x + y) * g + b   (in place on x), one row of 128 per block
// ---------------------------------------------------------------------------
__global__ __launch_bounds__(128) void ln_residual_kernel(
    float* __restrict__ x, const float* __restrict__ y,
    const float* __restrict__ g, const float* __restrict__ bta) {
  __shared__ float red[128];
  const int row = blockIdx.x;
  const int t   = threadIdx.x;
  const size_t off = (size_t)row * HDIM + t;
  const float v = x[off] + y[off];
  red[t] = v; __syncthreads();
  for (int s = 64; s > 0; s >>= 1) { if (t < s) red[t] += red[t + s]; __syncthreads(); }
  const float mu = red[0] * (1.0f / 128.0f);
  __syncthreads();
  const float d = v - mu;
  red[t] = d * d; __syncthreads();
  for (int s = 64; s > 0; s >>= 1) { if (t < s) red[t] += red[t + s]; __syncthreads(); }
  const float var = red[0] * (1.0f / 128.0f);
  x[off] = d * rsqrtf(var + 1e-5f) * g[t] + bta[t];
}

// ---------------------------------------------------------------------------
// Heads: xl = X[:, -1, :]; log_softmax(relu(xl@Wl1+bl1)@Wl2+bl2) and
// sigmoid(relu(xl@Wr1+br1)@Wr2+br2). One block per batch row (tiny).
// out = [level_logits (16*101)] ++ [rate (16)]
// ---------------------------------------------------------------------------
__global__ __launch_bounds__(128) void head_kernel(
    const float* __restrict__ X,
    const float* __restrict__ Wl1, const float* __restrict__ bl1,
    const float* __restrict__ Wl2, const float* __restrict__ bl2,
    const float* __restrict__ Wr1, const float* __restrict__ br1,
    const float* __restrict__ Wr2, const float* __restrict__ br2,
    float* __restrict__ out) {
  __shared__ float xl[128], hbuf[128], red[128];
  const int b = blockIdx.x;
  const int t = threadIdx.x;
  xl[t] = X[((size_t)b * SEQ + (SEQ - 1)) * HDIM + t];
  __syncthreads();
  float a1 = bl1[t], a2 = br1[t];
  for (int i = 0; i < 128; ++i) {
    a1 += xl[i] * Wl1[i * 128 + t];
    a2 += xl[i] * Wr1[i * 128 + t];
  }
  hbuf[t] = fmaxf(a1, 0.0f);
  __syncthreads();
  float lg = -1e30f;
  if (t < NCLS) {
    lg = bl2[t];
    for (int i = 0; i < 128; ++i) lg += hbuf[i] * Wl2[i * NCLS + t];
  }
  red[t] = lg; __syncthreads();
  for (int s = 64; s > 0; s >>= 1) { if (t < s) red[t] = fmaxf(red[t], red[t + s]); __syncthreads(); }
  const float mx = red[0];
  __syncthreads();
  red[t] = (t < NCLS) ? __expf(lg - mx) : 0.0f; __syncthreads();
  for (int s = 64; s > 0; s >>= 1) { if (t < s) red[t] += red[t + s]; __syncthreads(); }
  const float lse = mx + __logf(red[0]);
  if (t < NCLS) out[b * NCLS + t] = lg - lse;
  __syncthreads();
  red[t] = fmaxf(a2, 0.0f) * Wr2[t]; __syncthreads();
  for (int s = 64; s > 0; s >>= 1) { if (t < s) red[t] += red[t + s]; __syncthreads(); }
  if (t == 0) out[BSEQ * NCLS + b] = 1.0f / (1.0f + __expf(-(red[0] + br2[0])));
}

// ---------------------------------------------------------------------------
extern "C" void kernel_launch(void* const* d_in, const int* in_sizes, int n_in,
                              void* d_out, int out_size, void* d_ws, size_t ws_size,
                              hipStream_t stream) {
  const float* xq    = (const float*)d_in[0];
  const float* Wemb  = (const float*)d_in[1];
  const float* bemb  = (const float*)d_in[2];
  const float* Wqkv  = (const float*)d_in[3];
  const float* bqkv  = (const float*)d_in[4];
  const float* Wo    = (const float*)d_in[5];
  const float* bo    = (const float*)d_in[6];
  const float* W1    = (const float*)d_in[7];
  const float* b1    = (const float*)d_in[8];
  const float* W2    = (const float*)d_in[9];
  const float* b2    = (const float*)d_in[10];
  const float* ln1g  = (const float*)d_in[11];
  const float* ln1b  = (const float*)d_in[12];
  const float* ln2g  = (const float*)d_in[13];
  const float* ln2b  = (const float*)d_in[14];
  const float* Wl1   = (const float*)d_in[15];
  const float* bl1   = (const float*)d_in[16];
  const float* Wl2   = (const float*)d_in[17];
  const float* bl2   = (const float*)d_in[18];
  const float* Wr1   = (const float*)d_in[19];
  const float* br1   = (const float*)d_in[20];
  const float* Wr2   = (const float*)d_in[21];
  const float* br2   = (const float*)d_in[22];

  // workspace layout (floats): X | QKV | ATT | PROJ | FF1  = 80 MiB total
  float* X    = (float*)d_ws;
  float* QKV  = X    + (size_t)ROWS * HDIM;        //  8 MiB
  float* ATT  = QKV  + (size_t)ROWS * 384;         // 24 MiB
  float* PROJ = ATT  + (size_t)ROWS * HDIM;        //  8 MiB
  float* FF1  = PROJ + (size_t)ROWS * HDIM;        //  8 MiB (FF1: 32 MiB)

  embed_kernel<<<ROWS, 128, 0, stream>>>(xq, Wemb, bemb, X);

  for (int l = 0; l < NLAYER; ++l) {
    // QKV projection: 16384x128 @ 128x384
    gemm_wmma_kernel<128, 384, false>
        <<<dim3(384 / 128, ROWS / 128), 256, 0, stream>>>(
            X, Wqkv + (size_t)l * HDIM * 384, bqkv + (size_t)l * 384, QKV);
    // attention
    attn_kernel<<<dim3(SEQ / 128, NHEAD, BSEQ), 256, 0, stream>>>(QKV, ATT);
    // output projection: 16384x128 @ 128x128
    gemm_wmma_kernel<128, 128, false>
        <<<dim3(HDIM / 128, ROWS / 128), 256, 0, stream>>>(
            ATT, Wo + (size_t)l * HDIM * HDIM, bo + (size_t)l * HDIM, PROJ);
    ln_residual_kernel<<<ROWS, 128, 0, stream>>>(X, PROJ,
        ln1g + (size_t)l * HDIM, ln1b + (size_t)l * HDIM);
    // FFN: relu(x@W1+b1)@W2+b2
    gemm_wmma_kernel<128, 512, true>
        <<<dim3(512 / 128, ROWS / 128), 256, 0, stream>>>(
            X, W1 + (size_t)l * HDIM * 512, b1 + (size_t)l * 512, FF1);
    gemm_wmma_kernel<512, 128, false>
        <<<dim3(HDIM / 128, ROWS / 128), 256, 0, stream>>>(
            FF1, W2 + (size_t)l * 512 * HDIM, b2 + (size_t)l * HDIM, PROJ);
    ln_residual_kernel<<<ROWS, 128, 0, stream>>>(X, PROJ,
        ln2g + (size_t)l * HDIM, ln2b + (size_t)l * HDIM);
  }

  head_kernel<<<BSEQ, 128, 0, stream>>>(X, Wl1, bl1, Wl2, bl2,
                                        Wr1, br1, Wr2, br2, (float*)d_out);
}
